// NeRFRenderer_28578712387696
// MI455X (gfx1250) — matrix operand order repaired and checked by reading, not compiled, not run on hardware
//
#include <hip/hip_runtime.h>

#define NSTEPS 128
#define SALL   256
#define HID    64
#define HSTR   72   // LDS h-stage row stride in halfs: 144B rows keep 16B alignment

typedef __attribute__((ext_vector_type(2)))  float    v2f;
typedef __attribute__((ext_vector_type(8)))  float    v8f;
typedef __attribute__((ext_vector_type(16))) _Float16 v16h;

// ---- fast transcendentals: single hardware TRANS ops on CDNA5 ----
__device__ __forceinline__ float fexp_(float x) {
  return __builtin_amdgcn_exp2f(x * 1.4426950408889634f);   // v_exp_f32
}
__device__ __forceinline__ float flog_(float x) {
  return __builtin_amdgcn_logf(x) * 0.6931471805599453f;    // v_log_f32
}
__device__ __forceinline__ float tanh_(float x) {
#if __has_builtin(__builtin_amdgcn_tanhf)
  return __builtin_amdgcn_tanhf(x);                         // v_tanh_f32 (gfx1250)
#elif __has_builtin(__builtin_amdgcn_tanh_f32)
  return __builtin_amdgcn_tanh_f32(x);
#else
  float e = fexp_(2.0f * fabsf(x));                         // branch-free fallback
  float r = 1.0f - 2.0f / (e + 1.0f);
  return copysignf(r, x);
#endif
}
__device__ __forceinline__ float sigmoid_(float x)  { return 0.5f + 0.5f * tanh_(0.5f * x); }
__device__ __forceinline__ float softplus_(float x) { return fmaxf(x, 0.0f) + flog_(1.0f + fexp_(-fabsf(x))); }
__device__ __forceinline__ float clamp1_(float x)   { return fminf(fmaxf(x, -1.0f), 1.0f); }

// channel decode for the 31-way output reduction
__device__ __forceinline__ void decode_ch(int t, int& g, int& c) {
  c = 0;
  if      (t <  3) { g = 0;  c = t; }
  else if (t == 3) { g = 1; }
  else if (t == 4) { g = 2; }
  else if (t == 5) { g = 3; }
  else if (t == 6) { g = 4; }
  else if (t < 10) { g = 5;  c = t - 7;  }
  else if (t < 13) { g = 6;  c = t - 10; }
  else if (t < 16) { g = 7;  c = t - 13; }
  else if (t < 19) { g = 8;  c = t - 16; }
  else if (t < 22) { g = 9;  c = t - 19; }
  else if (t < 25) { g = 10; c = t - 22; }
  else if (t < 28) { g = 11; c = t - 25; }
  else             { g = 12; c = t - 28; }
}

// One 16-sample tile through: [16,K<=8] x [K,64] fp32 WMMA (bias fused via K=4
// homogeneous column) -> tanh -> fp16 LDS restage -> [16,64] x [64,16] f16 WMMA head.
template<bool WITHDIR>
__device__ __forceinline__ v8f mlp_tile(int lane, bool hi,
    float ox, float oy, float oz, float dx, float dy, float dz,
    const float* zsrc, const v2f* bA, const v2f* bB, const v16h* bHead,
    _Float16* hst)
{
  const int m = lane & 15;
  float zv = zsrc[m];
  float px = clamp1_(ox + dx * zv);
  float py = clamp1_(oy + dy * zv);
  float pz = clamp1_(oz + dz * zv);
  // A-frag for V_WMMA_F32_16X16X4_F32: lanes0-15 hold {K0,K1}, lanes16-31 {K2,K3}
  v2f a0, a1;
  if (hi) { a0[0] = pz; a0[1] = 1.0f; a1[0] = dz; a1[1] = 0.0f; }
  else    { a0[0] = px; a0[1] = py;   a1[0] = dx; a1[1] = dy;   }

  v8f hacc[4];
#pragma unroll
  for (int t = 0; t < 4; ++t) {
    v8f c = {0.f,0.f,0.f,0.f,0.f,0.f,0.f,0.f};
    c = __builtin_amdgcn_wmma_f32_16x16x4_f32(false, a0, false, bA[t], (short)0, c, false, false);
    if (WITHDIR)
      c = __builtin_amdgcn_wmma_f32_16x16x4_f32(false, a1, false, bB[t], (short)0, c, false, false);
    hacc[t] = c;
  }

  __syncthreads();   // protect h-stage reuse across successive tiles (uniform call sites)
#pragma unroll
  for (int t = 0; t < 4; ++t) {
#pragma unroll
    for (int r = 0; r < 8; ++r) {
      int M = r + (hi ? 8 : 0);                       // C layout: M = reg + 8*(lane>=16)
      hst[M * HSTR + t * 16 + m] = (_Float16)tanh_(hacc[t][r]);  // N = t*16 + lane%16
    }
  }
  __syncthreads();

  // Rebuild as f16 A-frags (ISA 16-bit A 16x32 layout), two K-groups of 32
  v16h ah0, ah1;
  const int kb = hi ? 8 : 0;
#pragma unroll
  for (int j = 0; j < 8; ++j) {
    ah0[j]     = hst[m * HSTR + kb + j];
    ah0[8 + j] = hst[m * HSTR + kb + 16 + j];
    ah1[j]     = hst[m * HSTR + 32 + kb + j];
    ah1[8 + j] = hst[m * HSTR + 32 + kb + 16 + j];
  }
  v8f c = {0.f,0.f,0.f,0.f,0.f,0.f,0.f,0.f};
  c = __builtin_amdgcn_wmma_f32_16x16x32_f16(false, ah0, false, bHead[0], (short)0, c, false, false);
  c = __builtin_amdgcn_wmma_f32_16x16x32_f16(false, ah1, false, bHead[1], (short)0, c, false, false);
  return c;
}

__global__ __launch_bounds__(256) void nerf_forward_kernel(
    const float* __restrict__ rays_o, const float* __restrict__ rays_d,
    const float* __restrict__ W1,  const float* __restrict__ b1,
    const float* __restrict__ Ws,  const float* __restrict__ Wp,
    const float* __restrict__ Wc1, const float* __restrict__ bc1,
    const float* __restrict__ Wrgb, const float* __restrict__ Wstyle,
    const float* __restrict__ Wback,
    float* __restrict__ out, int nrays)
{
  __shared__ float s_zc[NSTEPS], s_sc[NSTEPS], s_lc[NSTEPS];
  __shared__ float s_zf[NSTEPS], s_sf[NSTEPS], s_lf[NSTEPS];
  __shared__ float s_zm[SALL],  s_sm[SALL],  s_lm[SALL];
  __shared__ float s_w[SALL], s_wp[SALL], s_wp2[SALL], s_wp3[SALL], s_oz[SALL];
  __shared__ float s_col[9][SALL];             // rgb[0:3], style[3:6], back[6:9]
  __shared__ float s_part[8][32];              // per-wave partial channel sums
  __shared__ _Float16 s_h[8][16 * HSTR];       // per-wave fp16 activation stage

  const int ray = blockIdx.x;
  if (ray >= nrays) return;
  const int  tid  = threadIdx.x;
  const int  wave = tid >> 5;
  const int  lane = tid & 31;
  const bool hi   = lane >= 16;
  const int  nn   = lane & 15;

  const float ox = rays_o[ray*3+0], oy = rays_o[ray*3+1], oz = rays_o[ray*3+2];
  const float dx = rays_d[ray*3+0], dy = rays_d[ray*3+1], dz = rays_d[ray*3+2];

  // ---- near/far from aabb [-1,1]^3 ----
  float tnear, tfar;
  {
    float od[3] = {ox, oy, oz};
    float dd[3] = {dx, dy, dz};
    float tmin = -1e30f, tmax = 1e30f;
#pragma unroll
    for (int i = 0; i < 3; ++i) {
      float ds = dd[i];
      if (fabsf(ds) < 1e-9f) ds = 1e-9f;
      float inv = 1.0f / ds;
      float t1 = (-1.0f - od[i]) * inv;
      float t2 = ( 1.0f - od[i]) * inv;
      tmin = fmaxf(tmin, fminf(t1, t2));
      tmax = fminf(tmax, fmaxf(t1, t2));
    }
    tnear = fmaxf(tmin, 0.1f);
    tfar  = fmaxf(tmax, tnear + 1e-4f);
  }
  const float sample_dist = (tfar - tnear) / (float)NSTEPS;
  const float inv_range   = 1.0f / (tfar - tnear);

  // ---- per-lane constant weight fragments ----
  v2f bW1f[4], bC1a[4], bC1b[4];
#pragma unroll
  for (int t = 0; t < 4; ++t) {
    int n = t * 16 + nn;
    if (hi) { // K2,K3 rows: {W[2], bias} (bias column pairs with A's homogeneous 1)
      bW1f[t][0] = W1[2*HID + n];  bW1f[t][1] = b1[n];
      bC1a[t][0] = Wc1[2*HID + n]; bC1a[t][1] = bc1[n];
      bC1b[t][0] = Wc1[5*HID + n]; bC1b[t][1] = 0.0f;
    } else {  // K0,K1 rows
      bW1f[t][0] = W1[0*HID + n];  bW1f[t][1] = W1[1*HID + n];
      bC1a[t][0] = Wc1[0*HID + n]; bC1a[t][1] = Wc1[1*HID + n];
      bC1b[t][0] = Wc1[3*HID + n]; bC1b[t][1] = Wc1[4*HID + n];
    }
  }
  // f16 head B-frags: lanes0-15 carry K=kk*32+{0..7,16..23}, lanes16-31 carry K=kk*32+{8..15,24..31}
  v16h bHd[2], bHc[2];
#pragma unroll
  for (int kk = 0; kk < 2; ++kk) {
    int kb = kk * 32 + (hi ? 8 : 0);
#pragma unroll
    for (int j = 0; j < 16; ++j) {
      int k = kb + (j < 8 ? j : j + 8);
      float vd = (nn == 0) ? Ws[k] : ((nn == 1) ? Wp[k] : 0.0f);
      float vc = 0.0f;
      if      (nn < 3) vc = Wrgb[k*3 + nn];
      else if (nn < 6) vc = Wstyle[k*3 + (nn-3)];
      else if (nn < 9) vc = Wback[k*3 + (nn-6)];
      bHd[kk][j] = (_Float16)vd;
      bHc[kk][j] = (_Float16)vc;
    }
  }

  // ---- Phase 1: coarse stratified z ----
  if (tid < NSTEPS)
    s_zc[tid] = tnear + (tfar - tnear) * ((float)tid * (1.0f / (float)(NSTEPS - 1)));
  __syncthreads();

  // ---- Phase 2: coarse density MLP (wave w -> samples [16w,16w+16)) ----
  {
    v8f c = mlp_tile<false>(lane, hi, ox, oy, oz, dx, dy, dz, &s_zc[wave*16],
                            bW1f, bW1f, bHd, &s_h[wave][0]);
#pragma unroll
    for (int r = 0; r < 8; ++r) {
      int M = r + (hi ? 8 : 0);
      if      (nn == 0) s_sc[wave*16 + M] = softplus_(c[r]);
      else if (nn == 1) s_lc[wave*16 + M] = sigmoid_(c[r]);
    }
  }
  __syncthreads();

  // ---- Phase 3: deterministic inverse-CDF resampling ----
  // temp reuse (dead until phase 6): s_wp = alpha, s_w = z_mid, s_wp2 = raw cdf
  if (tid < NSTEPS) {
    float dzv = (tid < NSTEPS-1) ? (s_zc[tid+1] - s_zc[tid]) : sample_dist;
    s_wp[tid] = 1.0f - fexp_(-dzv * s_sc[tid]);
    if (tid < NSTEPS-1) s_w[tid] = 0.5f * (s_zc[tid] + s_zc[tid+1]);
  }
  __syncthreads();
  if (tid == 0) {  // only the inherently serial chains
    float T = 1.0f, cum = 0.0f;
    for (int i = 0; i < NSTEPS; ++i) {
      float a = s_wp[i];
      float wci = a * T;
      T *= (1.0f - a + 1e-15f);
      if (i >= 1 && i <= NSTEPS-2) { cum += wci + 1e-5f; s_wp2[i] = cum; } // raw cdf[i]
    }
    s_wp2[0] = 0.0f;
  }
  __syncthreads();
  if (tid < NSTEPS) {  // parallel inverse-CDF lookup (binary search on raw cdf)
    const float wsum = s_wp2[NSTEPS-2];
    const float invw = 1.0f / wsum;
    float u  = ((float)tid + 0.5f) * (1.0f / (float)NSTEPS);
    float uw = u * wsum;
    int lo = 0, hi2 = NSTEPS-1;            // upper_bound over 127 cdf entries
    while (lo < hi2) { int mid = (lo + hi2) >> 1; if (s_wp2[mid] <= uw) lo = mid + 1; else hi2 = mid; }
    int below = lo - 1; if (below > NSTEPS-2) below = NSTEPS-2; if (below < 0) below = 0;
    int above = lo;     if (above > NSTEPS-2) above = NSTEPS-2;
    float g0 = s_wp2[below] * invw, g1 = s_wp2[above] * invw;
    float den = g1 - g0; if (den < 1e-5f) den = 1.0f;
    float tt = (u - g0) / den;
    float z0 = s_w[below], z1 = s_w[above];
    s_zf[tid] = z0 + tt * (z1 - z0);
  }
  __syncthreads();

  // ---- Phase 4: fine density MLP ----
  {
    v8f c = mlp_tile<false>(lane, hi, ox, oy, oz, dx, dy, dz, &s_zf[wave*16],
                            bW1f, bW1f, bHd, &s_h[wave][0]);
#pragma unroll
    for (int r = 0; r < 8; ++r) {
      int M = r + (hi ? 8 : 0);
      if      (nn == 0) s_sf[wave*16 + M] = softplus_(c[r]);
      else if (nn == 1) s_lf[wave*16 + M] = sigmoid_(c[r]);
    }
  }
  __syncthreads();

  // ---- Phase 5: stable merge of two sorted 128-lists (== argsort of concat) ----
  if (tid == 0) {
    int i = 0, j = 0;
    for (int k = 0; k < SALL; ++k) {
      bool takeC = (j >= NSTEPS) || (i < NSTEPS && s_zc[i] <= s_zf[j]);
      if (takeC) { s_zm[k] = s_zc[i]; s_sm[k] = s_sc[i]; s_lm[k] = s_lc[i]; ++i; }
      else       { s_zm[k] = s_zf[j]; s_sm[k] = s_sf[j]; s_lm[k] = s_lf[j]; ++j; }
    }
  }
  __syncthreads();

  // ---- Phase 6: alphas in parallel, 4 cumprod scans on 4 threads ----
  {
    int i = tid;  // 0..255
    float dzv = (i < SALL-1) ? (s_zm[i+1] - s_zm[i]) : sample_dist;
    float sg = s_sm[i], lp = s_lm[i];
    s_w[i]   = 1.0f - fexp_(-dzv * sg);
    s_wp[i]  = 1.0f - fexp_(-dzv * sg * lp);
    s_wp2[i] = 1.0f - fexp_(-dzv * sg * (1.0f - lp));
    s_wp3[i] = 1.0f - fexp_(-dzv * lp);
    float ozv = (s_zm[i] - tnear) * inv_range;
    s_oz[i] = fminf(fmaxf(ozv, 0.0f), 1.0f);
  }
  __syncthreads();
  if (tid < 4) {
    float* arr = (tid == 0) ? s_w : (tid == 1) ? s_wp : (tid == 2) ? s_wp2 : s_wp3;
    float T = 1.0f;
    for (int i = 0; i < SALL; ++i) {
      float a = arr[i];
      arr[i] = a * T;
      T *= (1.0f - a + 1e-15f);
    }
  }
  __syncthreads();

  // ---- Phase 7: color MLP on merged samples (2 tiles per wave) ----
#pragma unroll
  for (int rep = 0; rep < 2; ++rep) {
    int tile = wave + rep * 8;
    v8f c = mlp_tile<true>(lane, hi, ox, oy, oz, dx, dy, dz, &s_zm[tile*16],
                           bC1a, bC1b, bHc, &s_h[wave][0]);
    if (nn < 9) {
#pragma unroll
      for (int r = 0; r < 8; ++r) {
        int M = r + (hi ? 8 : 0);
        s_col[nn][tile*16 + M] = sigmoid_(c[r]);
      }
    }
  }
  __syncthreads();

  // ---- Phase 8: 31-channel reduction, parallel over 8 waves ----
  {
    const float Yc3[3] = {0.8f, 1.0f, 0.0f};
    const float G = 0.71f;
    int g, c;
    decode_ch(lane, g, c);
    const float yc = Yc3[c];
    float acc = 0.0f;
    if (lane < 31) {
      const int i0 = wave * 32;
      for (int i = i0; i < i0 + 32; ++i) {
        float w = s_w[i], lp = s_lm[i];
        float v;
        switch (g) {
          case 0:  v = w * s_col[c][i]; break;
          case 1:  v = w * s_oz[i]; break;
          case 2:  v = w; break;
          case 3:  v = w * lp; break;
          case 4:  v = s_wp[i] + s_wp2[i]; break;
          case 5:  v = w * (lp * yc + (1.0f - lp) * G); break;
          case 6:  v = w * (lp * s_col[3+c][i] + (1.0f - lp) * G); break;
          case 7:  v = w * (lp * yc + (1.0f - lp) * s_col[6+c][i]); break;
          case 8:  v = s_wp[i] * yc + s_wp2[i] * G; break;
          case 9:  v = s_wp[i] * s_col[3+c][i] + s_wp2[i] * G; break;
          case 10: v = s_wp[i] * yc + s_wp2[i] * s_col[6+c][i]; break;
          case 11: { float p3 = s_wp3[i]; v = w * (p3 * yc + (1.0f - p3) * G); } break;
          default: { float p3 = s_wp3[i]; v = w * (p3 * s_col[3+c][i] + (1.0f - p3) * G); } break;
        }
        acc += v;
      }
    }
    s_part[wave][lane] = acc;
  }
  __syncthreads();
  if (tid < 31) {
    float acc = 0.0f;
#pragma unroll
    for (int wv = 0; wv < 8; ++wv) acc += s_part[wv][tid];
    out[(size_t)ray * 31 + tid] = acc;
  }
}

extern "C" void kernel_launch(void* const* d_in, const int* in_sizes, int n_in,
                              void* d_out, int out_size, void* d_ws, size_t ws_size,
                              hipStream_t stream) {
  (void)n_in; (void)out_size; (void)d_ws; (void)ws_size;
  const float* rays_o = (const float*)d_in[0];
  const float* rays_d = (const float*)d_in[1];
  const float* W1     = (const float*)d_in[2];
  const float* b1     = (const float*)d_in[3];
  const float* Ws     = (const float*)d_in[4];
  const float* Wp     = (const float*)d_in[5];
  const float* Wc1    = (const float*)d_in[6];
  const float* bc1    = (const float*)d_in[7];
  const float* Wrgb   = (const float*)d_in[8];
  const float* Wstyle = (const float*)d_in[9];
  const float* Wback  = (const float*)d_in[10];
  const int nrays = in_sizes[0] / 3;
  nerf_forward_kernel<<<nrays, 256, 0, stream>>>(
      rays_o, rays_d, W1, b1, Ws, Wp, Wc1, bc1, Wrgb, Wstyle, Wback,
      (float*)d_out, nrays);
}
